// MHA_17729624998342
// MI455X (gfx1250) — compile-verified, matrix operand order checked by reading, not run on hardware
//
#include <hip/hip_runtime.h>
#include <hip/hip_bf16.h>

// -----------------------------------------------------------------------------
// Fused causal MHA for MI455X (gfx1250, wave32, WMMA + async-to-LDS).
// bf16 inputs / fp32 accumulation via v_wmma_f32_16x16x32_bf16.
//   ws layout (bytes):
//     [0        ) xbf : x cast to bf16                [32768 x 1024]    67,108,864
//     [67108864 ) pbf : P_i cast+transposed           [16][192][1024]    6,291,456
//     [73400320 ) wbf : W_proj cast to bf16           [1024 x 1024]      2,097,152
//     [75497472 ) qk  : per-head Q|K (bf16)           [4096][128][128] 134,217,728
//     [209715200) vt  : per-head V transposed (bf16)  [4096][64][128]   67,108,864
//     [276824064) obf : attention out (bf16)          [32768 x 1024]    67,108,864
//   total ws = 343,932,928 bytes
// -----------------------------------------------------------------------------

typedef __bf16 bf16_t;
typedef __attribute__((ext_vector_type(16))) __bf16 bf16x16;
typedef __attribute__((ext_vector_type(8)))  __bf16 bf16x8;
typedef __attribute__((ext_vector_type(8)))  float  f32x8;
typedef __attribute__((ext_vector_type(4)))  int    i32x4;

#if __has_builtin(__builtin_amdgcn_global_load_async_to_lds_b128) && \
    __has_builtin(__builtin_amdgcn_s_wait_asynccnt)
#define USE_ASYNC 1
#else
#define USE_ASYNC 0
#endif

// Stage one 16-byte chunk global -> LDS. Async (ASYNCcnt) when available.
__device__ __forceinline__ void stage16(const bf16_t* __restrict__ g,
                                        bf16_t* __restrict__ l) {
#if USE_ASYNC
  __builtin_amdgcn_global_load_async_to_lds_b128(
      (__attribute__((address_space(1))) i32x4*)g,
      (__attribute__((address_space(3))) i32x4*)l, 0, 0);
#else
  *(bf16x8*)l = *(const bf16x8*)g;
#endif
}

// Wait for staged data (ASYNCcnt==0) then workgroup barrier.
__device__ __forceinline__ void stage_wait() {
#if USE_ASYNC
  __builtin_amdgcn_s_wait_asynccnt(0);
#endif
  __syncthreads();
}

__device__ __forceinline__ bf16_t f2bf(float f) {
  union { float f; unsigned u; } v; v.f = f;
  unsigned r = v.u + 0x7FFFu + ((v.u >> 16) & 1u);   // round-to-nearest-even
  union { unsigned short s; bf16_t b; } o; o.s = (unsigned short)(r >> 16);
  return o.b;
}

// Load a 16x32 bf16 fragment (A layout, or B^T layout where "row" = output col)
// from a K-contiguous row-major tile. Per ISA 7.12.2: lanes 0-15 hold row M=lane,
// K = 0..7 and 16..23; lanes 16-31 hold row M=lane-16, K = 8..15 and 24..31.
__device__ __forceinline__ bf16x16 load_frag(const bf16_t* __restrict__ base, int ld) {
  const int lane = threadIdx.x & 31;
  const int r    = lane & 15;
  const int koff = (lane >> 4) << 3;               // 0 or 8
  const bf16_t* p = base + r * ld + koff;
  bf16x8 lo = *(const bf16x8*)(p);                 // K = koff .. koff+7
  bf16x8 hi = *(const bf16x8*)(p + 16);            // K = koff+16 .. koff+23
  return __builtin_shufflevector(lo, hi, 0,1,2,3,4,5,6,7,8,9,10,11,12,13,14,15);
}

__device__ __forceinline__ f32x8 wmma_bf16(bf16x16 a, bf16x16 b, f32x8 c) {
  return __builtin_amdgcn_wmma_f32_16x16x32_bf16(false, a, false, b, (short)0, c,
                                                 false, false);
}

// ---------------------------------------------------------------- cast kernels
__global__ void cast_kernel(const float* __restrict__ src, bf16_t* __restrict__ dst,
                            size_t n) {
  size_t i = (size_t)blockIdx.x * blockDim.x + threadIdx.x;
  size_t stride = (size_t)gridDim.x * blockDim.x;
  for (; i < n; i += stride) dst[i] = f2bf(src[i]);
}

// dst[h][e][d] = P[h][d][e]  (h<16, e<192, d<1024) -> K-contiguous B^T layout
__global__ void castT_P_kernel(const float* __restrict__ P, bf16_t* __restrict__ dst) {
  const size_t n = 16ull * 192ull * 1024ull;
  size_t i = (size_t)blockIdx.x * blockDim.x + threadIdx.x;
  size_t stride = (size_t)gridDim.x * blockDim.x;
  for (; i < n; i += stride) {
    size_t h = i / (192ull * 1024ull);
    size_t rem = i - h * (192ull * 1024ull);
    size_t e = rem >> 10, d = rem & 1023ull;
    dst[i] = f2bf(P[(h * 1024ull + d) * 192ull + e]);
  }
}

// -------------------------------------------------------------- QKV projection
// One workgroup (128 thr / 4 waves) per (b,h): [128x1024]x[1024x192].
// Each wave owns 32 rows (2 A-frags reuse every B-frag). Double-buffered async LDS.
// Q,K (cols 0..127) -> qk[bh][128][128]; V (cols 128..191) -> vt[bh][64][128] (transposed).
__global__ __launch_bounds__(128) void qkv_kernel(const bf16_t* __restrict__ xbf,
                                                  const bf16_t* __restrict__ pbf,
                                                  bf16_t* __restrict__ qk,
                                                  bf16_t* __restrict__ vt) {
  const int bh = blockIdx.x;                // 0..4095
  const int b = bh >> 4, h = bh & 15;
  const int tid = threadIdx.x, wave = tid >> 5, lane = tid & 31;
  __shared__ __align__(16) bf16_t As[2][128 * 32];   // 2 x  8 KB
  __shared__ __align__(16) bf16_t Bs[2][192 * 32];   // 2 x 12 KB
  f32x8 acc0[12] = {};
  f32x8 acc1[12] = {};
  const bf16_t* xrow = xbf + (size_t)b * 128 * 1024;
  const bf16_t* prow = pbf + (size_t)h * 192 * 1024;

  auto stage = [&](int kk, int buf) {
    #pragma unroll
    for (int t = 0; t < 4; ++t) {                    // A: 128x32 = 512 chunks
      int i = (tid + t * 128) * 8;
      int r = i >> 5, c = i & 31;
      stage16(&xrow[(size_t)r * 1024 + kk + c], &As[buf][i]);
    }
    #pragma unroll
    for (int t = 0; t < 6; ++t) {                    // B^T: 192x32 = 768 chunks
      int i = (tid + t * 128) * 8;
      int r = i >> 5, c = i & 31;
      stage16(&prow[(size_t)r * 1024 + kk + c], &Bs[buf][i]);
    }
  };

  int buf = 0;
  stage(0, buf);
  stage_wait();
  for (int kk = 0; kk < 1024; kk += 32) {
    if (kk + 32 < 1024) stage(kk + 32, buf ^ 1);     // prefetch next tile
    const bf16_t* abase = &As[buf][wave * 32 * 32];
    bf16x16 af0 = load_frag(abase, 32);
    bf16x16 af1 = load_frag(abase + 16 * 32, 32);
    #pragma unroll
    for (int n = 0; n < 12; ++n) {
      bf16x16 bf = load_frag(&Bs[buf][n * 16 * 32], 32);
      acc0[n] = wmma_bf16(af0, bf, acc0[n]);
      acc1[n] = wmma_bf16(af1, bf, acc1[n]);
    }
    stage_wait();                                    // next tile ready, buf free
    buf ^= 1;
  }

  const int rhalf = lane >> 4, rlo = lane & 15;
  bf16_t* qkout = qk + (size_t)bh * 128 * 128;
  bf16_t* vtout = vt + (size_t)bh * 64 * 128;
  #pragma unroll
  for (int half = 0; half < 2; ++half) {
    const int r0 = wave * 32 + half * 16 + 8 * rhalf;
    #pragma unroll
    for (int n = 0; n < 12; ++n) {
      #pragma unroll
      for (int g = 0; g < 8; ++g) {
        float v = half ? acc1[n][g] : acc0[n][g];
        int row = r0 + g, col = n * 16 + rlo;
        if (n < 8) qkout[(size_t)row * 128 + col] = f2bf(v);            // Q|K
        else       vtout[(size_t)(col - 128) * 128 + row] = f2bf(v);    // V^T
      }
    }
  }
}

// ------------------------------------------------------------------- attention
// One workgroup (256 thr / 8 waves) per (b,h). S held in WMMA accumulators,
// softmax via shfl_xor reductions, probs through LDS, O = P.V.
__global__ __launch_bounds__(256) void attn_kernel(const bf16_t* __restrict__ qk,
                                                   const bf16_t* __restrict__ vt,
                                                   bf16_t* __restrict__ obf) {
  const int bh = blockIdx.x;
  const int b = bh >> 4, h = bh & 15;
  const int tid = threadIdx.x, wave = tid >> 5, lane = tid & 31;
  __shared__ __align__(16) bf16_t Ks[128 * 64];    // 16 KB, B^T for Q.K^T
  __shared__ __align__(16) bf16_t Vt[64 * 128];    // 16 KB, B^T for P.V
  __shared__ __align__(16) bf16_t Pb[128 * 128];   // 32 KB, softmax probs
  const bf16_t* qkbase = qk + (size_t)bh * 128 * 128;
  const bf16_t* vtbase = vt + (size_t)bh * 64 * 128;

  #pragma unroll
  for (int t = 0; t < 4; ++t) {                    // K rows: 128x64 = 1024 chunks
    int i = (tid + t * 256) * 8;
    int tt = i >> 6, e = i & 63;
    stage16(&qkbase[(size_t)tt * 128 + 64 + e], &Ks[i]);
  }
  #pragma unroll
  for (int t = 0; t < 4; ++t) {                    // V^T already transposed
    int i = (tid + t * 256) * 8;
    stage16(&vtbase[i], &Vt[i]);
  }
  stage_wait();

  // ---- phase 1: S = (Q K^T) * scale, causal mask, softmax -> Pb (bf16)
  f32x8 acc[8] = {};
  const bf16_t* qbase = qkbase + wave * 16 * 128;  // Q fragment straight from global
  bf16x16 q0 = load_frag(qbase, 128);              // e = 0..31
  bf16x16 q1 = load_frag(qbase + 32, 128);         // e = 32..63
  #pragma unroll
  for (int n = 0; n < 8; ++n) {
    acc[n] = wmma_bf16(q0, load_frag(&Ks[n * 16 * 64], 64), acc[n]);
    acc[n] = wmma_bf16(q1, load_frag(&Ks[n * 16 * 64 + 32], 64), acc[n]);
  }

  const float scale = 0.125f;                      // 1/sqrt(64)
  const int rhalf = lane >> 4, rlo = lane & 15;
  #pragma unroll
  for (int g = 0; g < 8; ++g) {
    const int row = wave * 16 + g + 8 * rhalf;
    float m = -3.0e38f;
    #pragma unroll
    for (int n = 0; n < 8; ++n) {
      int col = n * 16 + rlo;
      float v = acc[n][g] * scale;
      v = (col <= row) ? v : -3.0e38f;             // causal mask
      acc[n][g] = v;
      m = fmaxf(m, v);
    }
    #pragma unroll
    for (int off = 1; off < 16; off <<= 1) m = fmaxf(m, __shfl_xor(m, off, 32));
    float s = 0.f;
    #pragma unroll
    for (int n = 0; n < 8; ++n) {
      float p = __expf(acc[n][g] - m);
      acc[n][g] = p;
      s += p;
    }
    #pragma unroll
    for (int off = 1; off < 16; off <<= 1) s += __shfl_xor(s, off, 32);
    float inv = 1.0f / s;
    #pragma unroll
    for (int n = 0; n < 8; ++n)
      Pb[row * 128 + n * 16 + rlo] = f2bf(acc[n][g] * inv);
  }
  __syncthreads();

  // ---- phase 2: O = P (128x128) x V (128x64)
  f32x8 oacc[4] = {};
  #pragma unroll
  for (int ks = 0; ks < 128; ks += 32) {
    bf16x16 pfrag = load_frag(&Pb[wave * 16 * 128 + ks], 128);
    #pragma unroll
    for (int n = 0; n < 4; ++n)
      oacc[n] = wmma_bf16(pfrag, load_frag(&Vt[n * 16 * 128 + ks], 128), oacc[n]);
  }
  const int r0 = wave * 16 + 8 * rhalf;
  bf16_t* orow = obf + (size_t)b * 128 * 1024 + h * 64;   // [b][t][h*64+e]
  #pragma unroll
  for (int n = 0; n < 4; ++n)
    #pragma unroll
    for (int g = 0; g < 8; ++g)
      orow[(size_t)(r0 + g) * 1024 + n * 16 + rlo] = f2bf(oacc[n][g]);
}

// ----------------------------------------------------------- output projection
// y = O @ W^T + b.  W_proj row-major [n][d] is already the K-contiguous B^T.
// 128 thr / 4 waves; each wave owns 32 rows x 128 cols. Double-buffered async LDS.
__global__ __launch_bounds__(128) void proj_kernel(const bf16_t* __restrict__ obf,
                                                   const bf16_t* __restrict__ wbf,
                                                   const float* __restrict__ bias,
                                                   float* __restrict__ y) {
  const int mtile = blockIdx.x;   // 0..255
  const int ntile = blockIdx.y;   // 0..7
  const int tid = threadIdx.x, wave = tid >> 5, lane = tid & 31;
  __shared__ __align__(16) bf16_t As[2][128 * 32];
  __shared__ __align__(16) bf16_t Bs[2][128 * 32];
  f32x8 acc0[8] = {};
  f32x8 acc1[8] = {};
  const bf16_t* arow = obf + (size_t)mtile * 128 * 1024;
  const bf16_t* brow = wbf + (size_t)ntile * 128 * 1024;

  auto stage = [&](int kk, int buf) {
    #pragma unroll
    for (int t = 0; t < 4; ++t) {
      int i = (tid + t * 128) * 8;
      int r = i >> 5, c = i & 31;
      stage16(&arow[(size_t)r * 1024 + kk + c], &As[buf][i]);
      stage16(&brow[(size_t)r * 1024 + kk + c], &Bs[buf][i]);
    }
  };

  int buf = 0;
  stage(0, buf);
  stage_wait();
  for (int kk = 0; kk < 1024; kk += 32) {
    if (kk + 32 < 1024) stage(kk + 32, buf ^ 1);
    const bf16_t* abase = &As[buf][wave * 32 * 32];
    bf16x16 af0 = load_frag(abase, 32);
    bf16x16 af1 = load_frag(abase + 16 * 32, 32);
    #pragma unroll
    for (int n = 0; n < 8; ++n) {
      bf16x16 bf = load_frag(&Bs[buf][n * 16 * 32], 32);
      acc0[n] = wmma_bf16(af0, bf, acc0[n]);
      acc1[n] = wmma_bf16(af1, bf, acc1[n]);
    }
    stage_wait();
    buf ^= 1;
  }

  const int rhalf = lane >> 4, rlo = lane & 15;
  float* yrow = y + (size_t)mtile * 128 * 1024 + ntile * 128;
  #pragma unroll
  for (int half = 0; half < 2; ++half) {
    const int r0 = wave * 32 + half * 16 + 8 * rhalf;
    #pragma unroll
    for (int n = 0; n < 8; ++n) {
      float bv = bias[ntile * 128 + n * 16 + rlo];
      #pragma unroll
      for (int g = 0; g < 8; ++g) {
        float v = half ? acc1[n][g] : acc0[n][g];
        yrow[(size_t)(r0 + g) * 1024 + n * 16 + rlo] = v + bv;
      }
    }
  }
}

// ------------------------------------------------------------------------ glue
extern "C" void kernel_launch(void* const* d_in, const int* in_sizes, int n_in,
                              void* d_out, int out_size, void* d_ws, size_t ws_size,
                              hipStream_t stream) {
  (void)in_sizes; (void)n_in; (void)out_size; (void)ws_size;
  const float* x    = (const float*)d_in[0];   // [256,128,1024]
  const float* P    = (const float*)d_in[1];   // [16,1024,192]
  const float* W    = (const float*)d_in[2];   // [1024,1024]
  const float* bias = (const float*)d_in[3];   // [1024]
  float* y = (float*)d_out;

  char* ws = (char*)d_ws;
  bf16_t* xbf = (bf16_t*)(ws);
  bf16_t* pbf = (bf16_t*)(ws + 67108864);
  bf16_t* wbf = (bf16_t*)(ws + 73400320);
  bf16_t* qk  = (bf16_t*)(ws + 75497472);
  bf16_t* vt  = (bf16_t*)(ws + 209715200);
  bf16_t* obf = (bf16_t*)(ws + 276824064);

  cast_kernel<<<2048, 256, 0, stream>>>(x, xbf, 33554432ull);
  cast_kernel<<<512, 256, 0, stream>>>(W, wbf, 1048576ull);
  castT_P_kernel<<<1024, 256, 0, stream>>>(P, pbf);

  qkv_kernel<<<4096, 128, 0, stream>>>(xbf, pbf, qk, vt);
  attn_kernel<<<4096, 256, 0, stream>>>(qk, vt, obf);

  dim3 pg(256, 8);
  proj_kernel<<<pg, 128, 0, stream>>>(obf, wbf, bias, y);
}
